// ImageEncoding_60292750901342
// MI455X (gfx1250) — matrix operand order repaired and checked by reading, not compiled, not run on hardware
//
#include <hip/hip_runtime.h>

typedef __attribute__((ext_vector_type(16))) __bf16 v16bf;
typedef __attribute__((ext_vector_type(8)))  __bf16 v8bf;
typedef __attribute__((ext_vector_type(8)))  float  v8f;
typedef __attribute__((ext_vector_type(4)))  float  f4v;

#define GDIM 7
#define Bsz  8
#define Hsz  224
#define Wsz  224
#define Csz  256
#define NQ   64
#define TS   32            // 224/7
#define Ltot (TS*TS)       // 1024
#define LC   64            // chunk of L per iteration
#define NCH  (Ltot/LC)     // 16

// LDS pitches (in elements)
#define QP 272   // Q rows (ushort), 544B: 16B & 32B aligned fragments
#define KP 272   // K-chunk rows (ushort)
#define VP 80    // V-transposed rows (ushort), 160B
#define SP 68    // S rows (float), 272B
#define PP 80    // P rows (ushort), 160B

#define QS_OFF 0
#define KR_OFF (QS_OFF + 64*QP)                 // ushort elements
#define VT_OFF (KR_OFF + 64*KP)
#define S_BYTE_OFF  ((VT_OFF + 256*VP) * 2)     // bytes
#define P_BYTE_OFF  (S_BYTE_OFF + 64*SP*4)
#define A_BYTE_OFF  (P_BYTE_OFF + 64*PP*2)
#define L_BYTE_OFF  (A_BYTE_OFF + 64*4)
#define SMEM_BYTES  (L_BYTE_OFF + 64*4)         // ~138,752 B

static __device__ __forceinline__ unsigned short f2bf(float f) {
    unsigned u = __float_as_uint(f);
    u += 0x7FFFu + ((u >> 16) & 1u);   // round-to-nearest-even
    return (unsigned short)(u >> 16);
}

static __device__ __forceinline__ void st4bf(unsigned short* p, f4v v) {
    union { unsigned short s[4]; uint2 u; } t;
    t.s[0] = f2bf(v[0]); t.s[1] = f2bf(v[1]); t.s[2] = f2bf(v[2]); t.s[3] = f2bf(v[3]);
    *(uint2*)p = t.u;
}

static __device__ __forceinline__ v16bf ldA(const unsigned short* row, int k0, int h) {
    // A operand k order used by this code (not the ISA table order): lane half h holds K = {k0+h*16..+15} (contiguous, matches B)
    v8bf lo = *(const v8bf*)(row + k0 + h * 16);
    v8bf hi = *(const v8bf*)(row + k0 + h * 16 + 8);
    return __builtin_shufflevector(lo, hi, 0,1,2,3,4,5,6,7,8,9,10,11,12,13,14,15);
}

__global__ __launch_bounds__(256)
void ImageEncoding_attn_kernel(const float* __restrict__ images,
                               const float* __restrict__ queries,
                               float* __restrict__ out) {
    extern __shared__ char smem[];
    unsigned short* Qs = (unsigned short*)smem + QS_OFF;
    unsigned short* Kr = (unsigned short*)smem + KR_OFF;
    unsigned short* Vt = (unsigned short*)smem + VT_OFF;
    float*          Ss = (float*)(smem + S_BYTE_OFF);
    unsigned short* Pm = (unsigned short*)(smem + P_BYTE_OFF);
    float*          aL = (float*)(smem + A_BYTE_OFF);
    float*          lL = (float*)(smem + L_BYTE_OFF);

    const int tid  = threadIdx.x;
    const int lane = tid & 31;
    const int wave = tid >> 5;
    const int m    = lane & 15;     // row/col within fragment
    const int h    = lane >> 4;     // lane half
    const int blk  = blockIdx.x;
    const int b    = blk / 49;
    const int n    = blk % 49;
    const int ci   = n / 7, cj = n % 7;

    const int quarter = tid >> 6;
    const int t64     = tid & 63;

    // softmax quad mapping: 4 adjacent lanes per row (stay inside one wave32)
    const int srow_id = tid >> 2;   // 0..63
    const int seg     = tid & 3;    // 16 elements each

    // ---- load queries (64x256 f32) -> Qs bf16 ----
    for (int rr = 0; rr < 16; ++rr) {
        const int q = quarter * 16 + rr;
        const f4v v = ((const f4v*)(queries + q * Csz))[t64];
        st4bf(&Qs[q * QP + t64 * 4], v);
    }

    // per-row softmax running state (replicated across the quad)
    float m_run = -3.0e38f;
    float l_run = 0.0f;

    v8f accO[8];
    #pragma unroll
    for (int t = 0; t < 8; ++t) accO[t] = (v8f){0.f,0.f,0.f,0.f,0.f,0.f,0.f,0.f};

    const int mr  = wave & 3;        // row-tile of 4 (rows mr*16..)
    const int cc0 = (wave >> 2) * 8; // GEMM2 col-tile base (of 16)
    const int lc0 = wave >> 2;       // GEMM1 col tiles: lc0, lc0+2

    for (int ch = 0; ch < NCH; ++ch) {
        __syncthreads();   // previous GEMM2 done before overwriting K/V

        // ---- load K/V chunk (64 region rows x 256 ch) as bf16: row-major + transposed ----
        // images are streamed once (411MB > 192MB L2) -> non-temporal loads
        const int lbase = ch * LC;
        for (int rr = 0; rr < 16; ++rr) {
            const int lr = quarter * 16 + rr;
            const int l  = lbase + lr;
            const int ph = l >> 5, pw = l & 31;
            const f4v v = __builtin_nontemporal_load(
                (const f4v*)(images +
                    ((size_t)((b * Hsz + ci * TS + ph) * Wsz + cj * TS + pw)) * Csz) + t64);
            st4bf(&Kr[lr * KP + t64 * 4], v);
            const int c0 = t64 * 4;
            Vt[(c0 + 0) * VP + lr] = f2bf(v[0]);
            Vt[(c0 + 1) * VP + lr] = f2bf(v[1]);
            Vt[(c0 + 2) * VP + lr] = f2bf(v[2]);
            Vt[(c0 + 3) * VP + lr] = f2bf(v[3]);
        }
        __syncthreads();

        // ---- GEMM1: S(64xLC) = Q * K^T ----
        {
            v8f s0 = (v8f){0.f,0.f,0.f,0.f,0.f,0.f,0.f,0.f};
            v8f s1 = (v8f){0.f,0.f,0.f,0.f,0.f,0.f,0.f,0.f};
            #pragma unroll
            for (int ks = 0; ks < 8; ++ks) {
                const int k0 = ks * 32;
                v16bf a = ldA(&Qs[(mr * 16 + m) * QP], k0, h);
                v16bf b0 = *(const v16bf*)&Kr[(lc0 * 16 + m) * KP + k0 + h * 16];
                s0 = __builtin_amdgcn_wmma_f32_16x16x32_bf16(false, a, false, b0,
                                                             (short)0, s0, false, false);
                v16bf b1 = *(const v16bf*)&Kr[((lc0 + 2) * 16 + m) * KP + k0 + h * 16];
                s1 = __builtin_amdgcn_wmma_f32_16x16x32_bf16(false, a, false, b1,
                                                             (short)0, s1, false, false);
            }
            const int rbase = mr * 16 + h * 8;   // C layout: VGPR j -> M=j (+8 for high lanes)
            #pragma unroll
            for (int j = 0; j < 8; ++j) {
                Ss[(rbase + j) * SP + lc0 * 16 + m]       = s0[j];
                Ss[(rbase + j) * SP + (lc0 + 2) * 16 + m] = s1[j];
            }
        }
        __syncthreads();

        // ---- online softmax: quad of lanes per row, shuffle reductions ----
        {
            const float* srow = &Ss[srow_id * SP + seg * 16];
            float vals[16];
            #pragma unroll
            for (int i = 0; i < 4; ++i) {
                const f4v v = *(const f4v*)(srow + i * 4);
                vals[4*i+0] = v[0]; vals[4*i+1] = v[1];
                vals[4*i+2] = v[2]; vals[4*i+3] = v[3];
            }
            float cm = vals[0];
            #pragma unroll
            for (int i = 1; i < 16; ++i) cm = fmaxf(cm, vals[i]);
            cm = fmaxf(cm, __shfl_xor(cm, 1, 32));
            cm = fmaxf(cm, __shfl_xor(cm, 2, 32));

            const float m_new = fmaxf(m_run, cm);
            const float alpha = __expf(m_run - m_new);
            float s = 0.0f;
            union { unsigned short us[16]; uint4 u4[2]; } pk;
            #pragma unroll
            for (int i = 0; i < 16; ++i) {
                const float p = __expf(vals[i] - m_new);
                s += p;
                pk.us[i] = f2bf(p);
            }
            s += __shfl_xor(s, 1, 32);
            s += __shfl_xor(s, 2, 32);
            l_run = l_run * alpha + s;
            m_run = m_new;
            if (seg == 0) aL[srow_id] = alpha;
            uint4* pdst = (uint4*)&Pm[srow_id * PP + seg * 16];
            pdst[0] = pk.u4[0];
            pdst[1] = pk.u4[1];
        }
        __syncthreads();

        // ---- rescale accumulators, GEMM2: O += P * V ----
        {
            float af[8];
            #pragma unroll
            for (int j = 0; j < 8; ++j) {
                const float alo = aL[mr * 16 + j];
                const float ahi = aL[mr * 16 + j + 8];
                af[j] = h ? ahi : alo;
            }
            #pragma unroll
            for (int t = 0; t < 8; ++t)
                #pragma unroll
                for (int j = 0; j < 8; ++j) accO[t][j] *= af[j];

            #pragma unroll
            for (int ks = 0; ks < 2; ++ks) {
                const int k0 = ks * 32;
                v16bf a = ldA(&Pm[(mr * 16 + m) * PP], k0, h);
                #pragma unroll
                for (int t = 0; t < 8; ++t) {
                    v16bf bv = *(const v16bf*)&Vt[((cc0 + t) * 16 + m) * VP + k0 + h * 16];
                    accO[t] = __builtin_amdgcn_wmma_f32_16x16x32_bf16(false, a, false, bv,
                                                                      (short)0, accO[t],
                                                                      false, false);
                }
            }
        }
    }

    // ---- epilogue: divide by softmax denominator, store (write-once -> NT) ----
    if ((tid & 3) == 0) lL[srow_id] = l_run;
    __syncthreads();
    {
        float inv[8];
        #pragma unroll
        for (int j = 0; j < 8; ++j) {
            const float llo = lL[mr * 16 + j];
            const float lhi = lL[mr * 16 + j + 8];
            inv[j] = 1.0f / (h ? lhi : llo);
        }
        float* obase = out + (size_t)blk * NQ * Csz;
        #pragma unroll
        for (int t = 0; t < 8; ++t) {
            const int col = (cc0 + t) * 16 + m;
            #pragma unroll
            for (int j = 0; j < 8; ++j) {
                const int row = mr * 16 + h * 8 + j;
                __builtin_nontemporal_store(accO[t][j] * inv[j], &obase[row * Csz + col]);
            }
        }
    }
}

extern "C" void kernel_launch(void* const* d_in, const int* in_sizes, int n_in,
                              void* d_out, int out_size, void* d_ws, size_t ws_size,
                              hipStream_t stream) {
    const float* images  = (const float*)d_in[0];
    const float* queries = (const float*)d_in[1];
    float* out = (float*)d_out;

    hipFuncSetAttribute((const void*)ImageEncoding_attn_kernel,
                        hipFuncAttributeMaxDynamicSharedMemorySize, SMEM_BYTES);

    ImageEncoding_attn_kernel<<<Bsz * 49, 256, SMEM_BYTES, stream>>>(images, queries, out);
}
